// Operator_69337952026785
// MI455X (gfx1250) — compile-verified, matrix operand order checked by reading, not run on hardware
//
#include <hip/hip_runtime.h>

// ---------------- problem constants ----------------
constexpr int Bn = 4, Tn = 12, Cn = 16, Hn = 128, Wn = 192;
constexpr int HW   = Hn * Wn;        // 24576
constexpr int NCHW = Cn * HW;        // 393216
constexpr int TT   = Tn * Tn;        // 144
constexpr int NTAP = 9;              // 3x3 deformable kernel
constexpr int GRAMB = 48;            // blocks per batch for WMMA gram
constexpr int NCHUNK = NCHW / 32;    // 12288 K-chunks of 32

typedef _Float16 v16h __attribute__((ext_vector_type(16)));
typedef float    v8f  __attribute__((ext_vector_type(8)));

union V16H { v16h v; _Float16 h[16]; };
union V8F  { v8f  v; float    f[8];  };

// ---------------- x = x0 - mu (mu = last time slice) ----------------
__global__ void k_submu(const float* __restrict__ x0, float* __restrict__ X) {
    int idx = blockIdx.x * 256 + threadIdx.x;        // over B*T*NCHW
    int b = idx / (Tn * NCHW);
    int p = idx % NCHW;
    X[idx] = x0[idx] - x0[(b * Tn + (Tn - 1)) * NCHW + p];
}

// ---------------- WMMA Gram: partial[b,blk,(M,N)] = sum_K Q[b,M,K] P[b,N,K] ----
// A (16x32 f16) rows = Q time-rows (t), B (32x16 f16) cols = P rows (m),
// K loops over the C*H*W axis in chunks of 32, f32 accumulate.
// Per-lane layout per ISA 7.12.2: element e -> K = (e&7) + (e>>3)*16 + half*8,
// i.e. two contiguous 8-float runs at p0+half*8 and p0+16+half*8.
__global__ void __launch_bounds__(256) k_gram_wmma(
        const float* __restrict__ P, const float* __restrict__ Q,
        float* __restrict__ part) {
    int b = blockIdx.y;
    int wave = threadIdx.x >> 5, lane = threadIdx.x & 31;
    int half = lane >> 4;
    int row  = lane & 15;                    // t for A, m for B
    int rowc = (row < Tn) ? row : (Tn - 1);  // clamp pad rows (results unused)
    const float* qrow = Q + (size_t)(b * Tn + rowc) * NCHW;
    const float* prow = P + (size_t)(b * Tn + rowc) * NCHW;

    V8F acc;
#pragma unroll
    for (int r = 0; r < 8; r++) acc.f[r] = 0.f;

    int gw = blockIdx.x * 8 + wave;          // global wave id within b
    for (int ck = gw; ck < NCHUNK; ck += GRAMB * 8) {
        int p0 = ck * 32;
        V16H av, bv;
#pragma unroll
        for (int e = 0; e < 8; e++) {
            av.h[e]     = (_Float16)qrow[p0 + half * 8 + e];
            av.h[8 + e] = (_Float16)qrow[p0 + 16 + half * 8 + e];
            bv.h[e]     = (_Float16)prow[p0 + half * 8 + e];
            bv.h[8 + e] = (_Float16)prow[p0 + 16 + half * 8 + e];
        }
        acc.v = __builtin_amdgcn_wmma_f32_16x16x32_f16(
            false, av.v, false, bv.v, (short)0, acc.v, false, false);
    }

    // cross-wave reduce through LDS; write deterministic per-block partial tile
    __shared__ float tiles[8 * 256];
#pragma unroll
    for (int r = 0; r < 8; r++) {
        int M = r + half * 8, N = lane & 15;     // D: M = A-row(t), N = B-col(m)
        tiles[wave * 256 + M * 16 + N] = acc.f[r];
    }
    __syncthreads();
    int e = threadIdx.x;                         // 0..255 tile element
    float s = 0.f;
#pragma unroll
    for (int wv = 0; wv < 8; wv++) s += tiles[wv * 256 + e];
    part[((size_t)b * GRAMB + blockIdx.x) * 256 + e] = s;
}

// reduce the GRAMB partial tiles -> out[b*144 + t*12 + m]
__global__ void k_gram_red(const float* __restrict__ part, float* __restrict__ out) {
    int b = blockIdx.x;
    int e = threadIdx.x;                         // 0..255
    float s = 0.f;
    for (int blk = 0; blk < GRAMB; blk++)
        s += part[((size_t)b * GRAMB + blk) * 256 + e];
    int M = e >> 4, N = e & 15;
    if (M < Tn && N < Tn) out[b * TT + M * Tn + N] = s;
}

// ---------------- 12x12 symmetric eigensolve (Jacobi), sorted desc ----------------
__global__ void k_eig12(const float* __restrict__ G, float* __restrict__ VT,
                        float* __restrict__ SINV) {
    __shared__ float A[12][12], V[12][12];
    int b = blockIdx.x;
    if (threadIdx.x == 0) {
        for (int i = 0; i < 12; i++)
            for (int j = 0; j < 12; j++) {
                A[i][j] = G[b * TT + i * 12 + j];
                V[i][j] = (i == j) ? 1.f : 0.f;
            }
        for (int sweep = 0; sweep < 12; sweep++) {
            for (int p = 0; p < 11; p++)
                for (int q = p + 1; q < 12; q++) {
                    float apq = A[p][q];
                    if (fabsf(apq) < 1e-12f) continue;
                    float theta = (A[q][q] - A[p][p]) / (2.f * apq);
                    float tr = ((theta >= 0.f) ? 1.f : -1.f) /
                               (fabsf(theta) + sqrtf(1.f + theta * theta));
                    float c = rsqrtf(1.f + tr * tr);
                    float s = tr * c;
                    for (int k = 0; k < 12; k++) {
                        float akp = A[k][p], akq = A[k][q];
                        A[k][p] = c * akp - s * akq;
                        A[k][q] = s * akp + c * akq;
                    }
                    for (int k = 0; k < 12; k++) {
                        float apk = A[p][k], aqk = A[q][k];
                        A[p][k] = c * apk - s * aqk;
                        A[q][k] = s * apk + c * aqk;
                    }
                    for (int k = 0; k < 12; k++) {
                        float vkp = V[k][p], vkq = V[k][q];
                        V[k][p] = c * vkp - s * vkq;
                        V[k][q] = s * vkp + c * vkq;
                    }
                }
        }
        int idxv[12];
        float ev[12];
        for (int i = 0; i < 12; i++) { idxv[i] = i; ev[i] = A[i][i]; }
        for (int i = 0; i < 12; i++) {          // selection sort, descending
            int best = i;
            for (int j = i + 1; j < 12; j++)
                if (ev[idxv[j]] > ev[idxv[best]]) best = j;
            int tmp = idxv[i]; idxv[i] = idxv[best]; idxv[best] = tmp;
        }
        for (int m = 0; m < 12; m++) {
            int col = idxv[m];
            float e = ev[col];
            if (e < 1.f) e = 1.f;
            SINV[b * 12 + m] = rsqrtf(e);
            for (int t = 0; t < 12; t++) VT[b * TT + m * 12 + t] = V[t][col];
        }
    }
}

// ---------------- u[b,m,p] = sinv[b,m] * sum_t vt[b,m,t] x[b,t,p] ----------------
__global__ void k_proju(const float* __restrict__ X, const float* __restrict__ VT,
                        const float* __restrict__ SINV, float* __restrict__ U) {
    int b = blockIdx.y;
    int p = blockIdx.x * 256 + threadIdx.x;
    __shared__ float vt[TT], si[12];
    if (threadIdx.x < TT) vt[threadIdx.x] = VT[b * TT + threadIdx.x];
    if (threadIdx.x < 12) si[threadIdx.x] = SINV[b * 12 + threadIdx.x];
    __syncthreads();
    float xv[12];
#pragma unroll
    for (int t = 0; t < 12; t++) xv[t] = X[(size_t)(b * Tn + t) * NCHW + p];
#pragma unroll
    for (int m = 0; m < 12; m++) {
        float s = 0.f;
#pragma unroll
        for (int t = 0; t < 12; t++) s += vt[m * 12 + t] * xv[t];
        U[(size_t)(b * Tn + m) * NCHW + p] = s * si[m];
    }
}

// ---------------- a[b] = svt[b] @ MW^T + MB  (144x144, tiny) ----------------
__global__ void k_mixer(const float* __restrict__ SVT, const float* __restrict__ MW,
                        const float* __restrict__ MB, float* __restrict__ Amat) {
    int b = blockIdx.x;
    __shared__ float sv[TT];
    if (threadIdx.x < TT) sv[threadIdx.x] = SVT[b * TT + threadIdx.x];
    __syncthreads();
    int r = threadIdx.x;
    if (r < TT) {
        float s = MB[r];
        for (int k = 0; k < TT; k++) s += sv[k] * MW[r * TT + k];
        Amat[b * TT + r] = s;
    }
}

// ---------------- u2 = a·u ; off = u2·off_p  (fused) ----------------
__global__ void k_u2off(const float* __restrict__ U, const float* __restrict__ Amat,
                        const float* __restrict__ OffP, float* __restrict__ XS,
                        float* __restrict__ OFF) {
    int bt = blockIdx.y;
    int b = bt / Tn, t = bt % Tn;
    int p = blockIdx.x * 256 + threadIdx.x;
    __shared__ float av[12];
    __shared__ float op[2 * NTAP * Cn];            // 288
    if (threadIdx.x < 12) av[threadIdx.x] = Amat[b * TT + t * 12 + threadIdx.x];
    for (int s = threadIdx.x; s < 2 * NTAP * Cn; s += 256) {
        int q = s / Cn, c = s % Cn;
        op[s] = OffP[(q * Tn + t) * Cn + c];       // off_p[layer, q, t, c]
    }
    __syncthreads();
    float u2[Cn];
#pragma unroll
    for (int c = 0; c < Cn; c++) {
        float s = 0.f;
#pragma unroll
        for (int m = 0; m < 12; m++)
            s += av[m] * U[(size_t)((b * Tn + m) * Cn + c) * HW + p];
        u2[c] = s;
        XS[(size_t)(bt * Cn + c) * HW + p] = s;
    }
#pragma unroll
    for (int q = 0; q < 2 * NTAP; q++) {
        float s = 0.f;
#pragma unroll
        for (int c = 0; c < Cn; c++) s += u2[c] * op[q * Cn + c];
        OFF[(size_t)(bt * 2 * NTAP + q) * HW + p] = s;
    }
}

// ---------------- WMMA deformable conv + residual ----------------
__global__ void __launch_bounds__(256) k_deform(
        const float* __restrict__ XIN, const float* __restrict__ OFF,
        const float* __restrict__ CW, const float* __restrict__ CB,
        float* __restrict__ XOUT) {
    __shared__ _Float16 wsh[NTAP * 256];           // [tap][o*16+c] as f16
    __shared__ float bsh[Cn];
    for (int s = threadIdx.x; s < NTAP * 256; s += 256) {
        int j = s >> 8, oc = s & 255;
        wsh[j * 256 + oc] = (_Float16)CW[oc * NTAP + j];
    }
    if (threadIdx.x < Cn) bsh[threadIdx.x] = CB[threadIdx.x];
    __syncthreads();

    int wave = threadIdx.x >> 5;
    int lane = threadIdx.x & 31;
    int tile = blockIdx.x * 8 + wave;              // 16 pixels per tile
    int idx0 = tile * 16;                          // into bt*H*W (W%16==0)
    int bt  = idx0 / HW;
    int rem = idx0 % HW;
    int h   = rem / Wn;
    int w0  = rem % Wn;
    int px   = lane & 15;                          // N (pixel) index
    int half = lane >> 4;                          // 0: ch 0-7, 1: ch 8-15
    int w    = w0 + px;

    const float* xin_b = XIN + (size_t)bt * NCHW;
    const float* off_b = OFF + (size_t)bt * 2 * NTAP * HW + h * Wn + w;
    __builtin_prefetch(xin_b + half * 8 * HW + rem, 0, 1);

    // accumulator: bias + residual (C/D layout: VGPR r <-> M = r + half*8)
    V8F acc;
#pragma unroll
    for (int r = 0; r < 8; r++) {
        int o = r + half * 8;
        acc.f[r] = bsh[o] + xin_b[o * HW + h * Wn + w];
    }

#pragma unroll
    for (int m = 0; m < 5; m++) {                  // tap pairs (0,1)..(8,-)
        int j0 = 2 * m, j1 = 2 * m + 1;
        V16H av, bv;
        // A: element e -> tap = e>>3, c = half*8 + (e&7), row o = lane&15
        int o = lane & 15;
#pragma unroll
        for (int e = 0; e < 16; e++) {
            int tap = (e >> 3) ? j1 : j0;
            int c = half * 8 + (e & 7);
            av.h[e] = (tap < NTAP) ? wsh[tap * 256 + o * Cn + c] : (_Float16)0.f;
        }
        // B: bilinear samples, same (tap, c) indexing, N = px via lane
#pragma unroll
        for (int tp = 0; tp < 2; tp++) {
            int j = tp ? j1 : j0;
            if (j >= NTAP) {                       // uniform branch (m==4)
#pragma unroll
                for (int cc = 0; cc < 8; cc++) bv.h[tp * 8 + cc] = (_Float16)0.f;
                continue;
            }
            int ky = j / 3, kx = j % 3;
            float offy = off_b[(2 * j) * HW];
            float offx = off_b[(2 * j + 1) * HW];
            float ysf = (float)(h + ky - 1) + offy;
            float xsf = (float)(w + kx - 1) + offx;
            float y0f = floorf(ysf), x0f = floorf(xsf);
            float wy = ysf - y0f, wx = xsf - x0f;
            int y0 = (int)y0f, x0 = (int)x0f;
            int y1 = y0 + 1, x1 = x0 + 1;
            float w00 = (1.f - wy) * (1.f - wx), w01 = (1.f - wy) * wx;
            float w10 = wy * (1.f - wx),         w11 = wy * wx;
            bool vy0 = (y0 >= 0) && (y0 < Hn), vy1 = (y1 >= 0) && (y1 < Hn);
            bool vx0 = (x0 >= 0) && (x0 < Wn), vx1 = (x1 >= 0) && (x1 < Wn);
            w00 = (vy0 && vx0) ? w00 : 0.f;
            w01 = (vy0 && vx1) ? w01 : 0.f;
            w10 = (vy1 && vx0) ? w10 : 0.f;
            w11 = (vy1 && vx1) ? w11 : 0.f;
            int y0c = min(max(y0, 0), Hn - 1), y1c = min(max(y1, 0), Hn - 1);
            int x0c = min(max(x0, 0), Wn - 1), x1c = min(max(x1, 0), Wn - 1);
            int s00 = y0c * Wn + x0c, s01 = y0c * Wn + x1c;
            int s10 = y1c * Wn + x0c, s11 = y1c * Wn + x1c;
#pragma unroll
            for (int cc = 0; cc < 8; cc++) {
                int c = half * 8 + cc;
                const float* xc = xin_b + c * HW;
                float sv = w00 * xc[s00] + w01 * xc[s01] +
                           w10 * xc[s10] + w11 * xc[s11];
                bv.h[tp * 8 + cc] = (_Float16)sv;
            }
        }
        acc.v = __builtin_amdgcn_wmma_f32_16x16x32_f16(
            false, av.v, false, bv.v, (short)0, acc.v, false, false);
    }

#pragma unroll
    for (int r = 0; r < 8; r++) {
        int o = r + half * 8;
        XOUT[(size_t)bt * NCHW + o * HW + h * Wn + w] = acc.f[r];
    }
}

// ---------------- a_sel[b,m] = svt[b] @ select_w^T + select_b ----------------
__global__ void k_select(const float* __restrict__ SVT, const float* __restrict__ SW,
                         const float* __restrict__ SB, float* __restrict__ ASel) {
    int b = blockIdx.x;
    __shared__ float sv[TT];
    for (int i = threadIdx.x; i < TT; i += 32) sv[i] = SVT[b * TT + i];
    __syncthreads();
    int m = threadIdx.x;
    if (m < 12) {
        float s = SB[m];
        for (int k = 0; k < TT; k++) s += sv[k] * SW[m * TT + k];
        ASel[b * 12 + m] = s;
    }
}

// ---------------- out = sum_m a_sel[b,m] * u[b,m] + mu ----------------
__global__ void k_final(const float* __restrict__ U, const float* __restrict__ ASel,
                        const float* __restrict__ x0, float* __restrict__ out) {
    int b = blockIdx.y;
    int p = blockIdx.x * 256 + threadIdx.x;
    __shared__ float av[12];
    if (threadIdx.x < 12) av[threadIdx.x] = ASel[b * 12 + threadIdx.x];
    __syncthreads();
    float s = x0[(size_t)(b * Tn + (Tn - 1)) * NCHW + p];   // mu
#pragma unroll
    for (int m = 0; m < 12; m++)
        s += av[m] * U[(size_t)(b * Tn + m) * NCHW + p];
    out[(size_t)b * NCHW + p] = s;
}

// =====================================================================
extern "C" void kernel_launch(void* const* d_in, const int* in_sizes, int n_in,
                              void* d_out, int out_size, void* d_ws, size_t ws_size,
                              hipStream_t stream) {
    const float* x0       = (const float*)d_in[0];
    const float* mixer_w  = (const float*)d_in[1];
    const float* mixer_b  = (const float*)d_in[2];
    const float* select_w = (const float*)d_in[3];
    const float* select_b = (const float*)d_in[4];
    const float* cnn_w    = (const float*)d_in[5];
    const float* cnn_b    = (const float*)d_in[6];
    const float* off_p    = (const float*)d_in[7];

    float* ws = (float*)d_ws;
    const size_t BIG = (size_t)Bn * Tn * NCHW;          // 18,874,368
    float* P0   = ws;
    float* P1   = P0 + BIG;
    float* P2   = P1 + BIG;
    float* Ubuf = P2 + BIG;
    float* OFFb = Ubuf + BIG;                           // B*T*18*HW
    float* G    = OFFb + (size_t)Bn * Tn * 2 * NTAP * HW;
    float* VT   = G + Bn * TT;
    float* SINV = VT + Bn * TT;
    float* SVT  = SINV + Bn * Tn;
    float* AM   = SVT + Bn * TT;
    float* AS   = AM + Bn * TT;
    float* GP   = AS + Bn * Tn;                         // gram partials 4*48*256
    (void)in_sizes; (void)n_in; (void)out_size; (void)ws_size;

    dim3 gP(NCHW / 256, Bn);              // 1536 x 4
    dim3 gU(HW / 256, Bn * Tn);           // 96 x 48
    dim3 gG(GRAMB, Bn);                   // 48 x 4
    const int dcBlocks = (Bn * Tn * HW) / (16 * 8);     // 9216

    k_submu<<<(Bn * Tn * NCHW) / 256, 256, 0, stream>>>(x0, P0);

    float* cur = P0;
    for (int i = 0; i < 5; i++) {
        k_gram_wmma<<<gG, 256, 0, stream>>>(cur, cur, GP);
        k_gram_red<<<Bn, 256, 0, stream>>>(GP, G);
        k_eig12<<<Bn, 32, 0, stream>>>(G, VT, SINV);
        k_proju<<<gP, 256, 0, stream>>>(cur, VT, SINV, Ubuf);
        k_gram_wmma<<<gG, 256, 0, stream>>>(Ubuf, cur, GP);  // svt[b,t,m]=u[m]·x[t]
        k_gram_red<<<Bn, 256, 0, stream>>>(GP, SVT);
        k_mixer<<<Bn, 256, 0, stream>>>(SVT, mixer_w + (size_t)i * TT * TT,
                                        mixer_b + (size_t)i * TT, AM);
        float *s0, *s1;
        if (cur == P0)      { s0 = P1; s1 = P2; }
        else if (cur == P1) { s0 = P0; s1 = P2; }
        else                { s0 = P0; s1 = P1; }
        k_u2off<<<gU, 256, 0, stream>>>(Ubuf, AM,
                                        off_p + (size_t)i * 2 * NTAP * Tn * Cn,
                                        s0, OFFb);
        // xs = dc(xs, w[i]) + xs ; xs = dc(xs, w[i+1]) + xs
        k_deform<<<dcBlocks, 256, 0, stream>>>(
            s0, OFFb, cnn_w + (size_t)i * Cn * Cn * NTAP,
            cnn_b + (size_t)i * Cn, s1);
        k_deform<<<dcBlocks, 256, 0, stream>>>(
            s1, OFFb, cnn_w + (size_t)(i + 1) * Cn * Cn * NTAP,
            cnn_b + (size_t)(i + 1) * Cn, s0);
        cur = s0;
    }

    // final POD + selection
    k_gram_wmma<<<gG, 256, 0, stream>>>(cur, cur, GP);
    k_gram_red<<<Bn, 256, 0, stream>>>(GP, G);
    k_eig12<<<Bn, 32, 0, stream>>>(G, VT, SINV);
    k_proju<<<gP, 256, 0, stream>>>(cur, VT, SINV, Ubuf);
    k_gram_wmma<<<gG, 256, 0, stream>>>(Ubuf, cur, GP);
    k_gram_red<<<Bn, 256, 0, stream>>>(GP, SVT);
    k_select<<<Bn, 32, 0, stream>>>(SVT, select_w, select_b, AS);
    k_final<<<gP, 256, 0, stream>>>(Ubuf, AS, x0, (float*)d_out);
}
